// NeuronQwen3MoeDecoderLayer_49864570306614
// MI455X (gfx1250) — compile-verified, matrix operand order checked by reading, not run on hardware
//
#include <hip/hip_runtime.h>

// ---------------------------------------------------------------------------
// Problem constants (from reference)
// ---------------------------------------------------------------------------
constexpr int S_  = 2048;   // sequence length (B = 1)
constexpr int D_  = 2048;   // model dim
constexpr int H_  = 32;     // query heads
constexpr int HK_ = 4;      // kv heads
constexpr int HD_ = 128;    // head dim
constexpr int E_  = 32;     // experts
constexpr int TOPK_ = 4;    // top-k experts
constexpr int F_  = 768;    // expert ffn dim
constexpr float EPS_ = 1e-6f;

// ---------------------------------------------------------------------------
// WMMA types / helpers
// ---------------------------------------------------------------------------
typedef __attribute__((ext_vector_type(16))) __bf16 v16bf;
typedef __attribute__((ext_vector_type(8)))  float  v8f;
typedef __attribute__((ext_vector_type(4)))  unsigned uint32x4_;
typedef __attribute__((ext_vector_type(8)))  int      int32x8_;
typedef __attribute__((ext_vector_type(4)))  int      int32x4_;

union FragBF {
    v16bf v;
    uint4 q[2];
};

__device__ __forceinline__ unsigned f2bf_u(float f) {
    union { float f; unsigned u; } x;
    x.f = f;
    unsigned u = x.u;
    u += 0x7fffu + ((u >> 16) & 1u);   // round-to-nearest-even
    return u >> 16;
}
__device__ __forceinline__ unsigned short f2bf(float f) {
    return (unsigned short)f2bf_u(f);
}
// pack two floats into one dword of bf16 (lo = a, hi = b)
__device__ __forceinline__ unsigned pk2(float a, float b) {
    return f2bf_u(a) | (f2bf_u(b) << 16);
}

// convert 8 consecutive floats -> 8 bf16, one ds_store_b128 (zero-fill if !ok)
__device__ __forceinline__ void store_pk8(unsigned short* dst, const float* src, bool ok) {
    uint4 pk = make_uint4(0u, 0u, 0u, 0u);
    if (ok) {
        float4 f0 = *(const float4*)src;
        float4 f1 = *(const float4*)(src + 4);
        pk.x = pk2(f0.x, f0.y);
        pk.y = pk2(f0.z, f0.w);
        pk.z = pk2(f1.x, f1.y);
        pk.w = pk2(f1.z, f1.w);
    }
    *(uint4*)dst = pk;
}

// Stage a 32(k) x 128(n) fp32 tile of B into LDS transposed [n][k] bf16.
// 256 threads: each handles 4 k x 4 n, emitting 4 ds_store_b64.
__device__ __forceinline__ void stage_bT(unsigned short* Bs, const float* B,
                                         int ldb, int tid) {
    int kq = tid >> 5;            // 0..7  -> k base 4*kq
    int n0 = (tid & 31) * 4;      // 0..124
    const float* p = B + (size_t)(4 * kq) * ldb + n0;
    float4 r0 = *(const float4*)(p);
    float4 r1 = *(const float4*)(p + ldb);
    float4 r2 = *(const float4*)(p + 2 * (size_t)ldb);
    float4 r3 = *(const float4*)(p + 3 * (size_t)ldb);
    unsigned short* b = Bs + n0 * 32 + 4 * kq;
    *(uint2*)(b + 0 * 32) = make_uint2(pk2(r0.x, r1.x), pk2(r2.x, r3.x));
    *(uint2*)(b + 1 * 32) = make_uint2(pk2(r0.y, r1.y), pk2(r2.y, r3.y));
    *(uint2*)(b + 2 * 32) = make_uint2(pk2(r0.z, r1.z), pk2(r2.z, r3.z));
    *(uint2*)(b + 3 * 32) = make_uint2(pk2(r0.w, r1.w), pk2(r2.w, r3.w));
}

// ---------------------------------------------------------------------------
// Tensor Data Mover: 1-D copy of one pre-tiled 8KB bf16 B tile into LDS.
// D# built per CDNA5 ISA: count=1, type=2 ("image"),
// data_size=2B, tile_dim0=4096 elems, tensor_dim0=4096, stride=4096.
// clang-23 toolchain: 6-arg builtin (g0, g1, g2, g3, g4, cpol).
// ---------------------------------------------------------------------------
__device__ __forceinline__ void tdm_load_tile_8kb(unsigned lds_off, const void* gsrc) {
    unsigned long long ga = (unsigned long long)(size_t)gsrc;
    uint32x4_ g0;
    g0[0] = 1u;                                                // count=1
    g0[1] = lds_off;                                           // lds_addr (bytes)
    g0[2] = (unsigned)(ga & 0xffffffffu);                      // global_addr[31:0]
    g0[3] = (unsigned)((ga >> 32) & 0x01ffffffu) | (2u << 30); // addr[56:32] | type=2
    int32x8_ g1;
    g1[0] = (int)(1u << 16);        // workgroup_mask=0, data_size=1 (2 bytes)
    g1[1] = (int)(4096u << 16);     // tensor_dim0[15:0] = 4096
    g1[2] = (int)(1u << 16);        // tensor_dim0[31:16]=0, tensor_dim1[15:0]=1
    g1[3] = (int)(4096u << 16);     // tensor_dim1[31:16]=0, tile_dim0=4096
    g1[4] = 0;                      // tile_dim1=0 (unused), tile_dim2=0
    g1[5] = 4096;                   // tensor_dim0_stride[31:0]
    g1[6] = 0;                      // stride hi, tensor_dim1_stride lo
    g1[7] = 0;
    int32x4_ z4;
    z4[0] = 0; z4[1] = 0; z4[2] = 0; z4[3] = 0;
    int32x8_ z8;
    z8[0] = 0; z8[1] = 0; z8[2] = 0; z8[3] = 0;
    z8[4] = 0; z8[5] = 0; z8[6] = 0; z8[7] = 0;
    __builtin_amdgcn_tensor_load_to_lds(g0, g1, z4, z4, z8, 0);
}

#define WMMA_BF16(A, B, C) \
    __builtin_amdgcn_wmma_f32_16x16x32_bf16(false, (A), false, (B), (short)0, (C), false, false)

// load the two b128 halves of a 16x32 bf16 fragment row from LDS
#define LOAD_FRAG(frag, base, rowoff)                                     \
    do {                                                                  \
        (frag).q[0] = *(const uint4*)&(base)[(rowoff) + half * 8];        \
        (frag).q[1] = *(const uint4*)&(base)[(rowoff) + 16 + half * 8];   \
    } while (0)

// ---------------------------------------------------------------------------
// Weight pre-tiling: W (K x N fp32, row-major) -> Bt tiles
// [(N/128)][(K/32)][128n][32k] bf16 ; each tile 8KB contiguous, fragment order
// ---------------------------------------------------------------------------
__global__ void tile_weights_kernel(const float* __restrict__ W,
                                    unsigned short* __restrict__ Bt,
                                    int K, int N) {
    int tk = blockIdx.x;           // K/32 tiles
    int tn = blockIdx.y;           // N/128 tiles
    int tid = threadIdx.x;
    int kq = tid >> 5;             // k base 4*kq
    int n0 = (tid & 31) * 4;
    const float* p = W + (size_t)(tk * 32 + 4 * kq) * N + tn * 128 + n0;
    float4 r0 = *(const float4*)(p);
    float4 r1 = *(const float4*)(p + N);
    float4 r2 = *(const float4*)(p + 2 * (size_t)N);
    float4 r3 = *(const float4*)(p + 3 * (size_t)N);
    unsigned short* b = Bt + ((size_t)tn * gridDim.x + tk) * 4096 + n0 * 32 + 4 * kq;
    *(uint2*)(b + 0 * 32) = make_uint2(pk2(r0.x, r1.x), pk2(r2.x, r3.x));
    *(uint2*)(b + 1 * 32) = make_uint2(pk2(r0.y, r1.y), pk2(r2.y, r3.y));
    *(uint2*)(b + 2 * 32) = make_uint2(pk2(r0.z, r1.z), pk2(r2.z, r3.z));
    *(uint2*)(b + 3 * 32) = make_uint2(pk2(r0.w, r1.w), pk2(r2.w, r3.w));
}

// ---------------------------------------------------------------------------
// RMSNorm over rows of length `dim`
// ---------------------------------------------------------------------------
__global__ void rmsnorm_kernel(const float* __restrict__ X,
                               const float* __restrict__ W,
                               float* __restrict__ Y, int dim) {
    int row = blockIdx.x;
    int tid = threadIdx.x;
    const float* x = X + (size_t)row * dim;
    float ss = 0.f;
    for (int d = tid; d < dim; d += 256) { float v = x[d]; ss += v * v; }
    #pragma unroll
    for (int o = 16; o > 0; o >>= 1) ss += __shfl_xor(ss, o, 32);
    __shared__ float red[8];
    int wid = tid >> 5, lane = tid & 31;
    if (lane == 0) red[wid] = ss;
    __syncthreads();
    float tot = 0.f;
    #pragma unroll
    for (int w = 0; w < 8; ++w) tot += red[w];
    float rs = rsqrtf(tot / (float)dim + EPS_);
    float* y = Y + (size_t)row * dim;
    for (int d = tid; d < dim; d += 256) y[d] = x[d] * rs * W[d];
}

// ---------------------------------------------------------------------------
// Per-(token, head) RMSNorm + RoPE, in place.  block = 128 threads (one row)
// ---------------------------------------------------------------------------
__global__ void qknorm_rope_kernel(float* __restrict__ X,
                                   const float* __restrict__ W,
                                   const int* __restrict__ pos,
                                   int nheads) {
    int t = blockIdx.x, h = blockIdx.y, d = threadIdx.x;
    size_t base = ((size_t)t * nheads + h) * HD_;
    float val = X[base + d];
    int   pd  = (d < 64) ? d + 64 : d - 64;
    float oth = X[base + pd];

    float ss = val * val;
    #pragma unroll
    for (int o = 16; o > 0; o >>= 1) ss += __shfl_xor(ss, o, 32);
    __shared__ float red[4];
    int wid = d >> 5, lane = d & 31;
    if (lane == 0) red[wid] = ss;
    __syncthreads();                       // also fences the global reads above
    float tot = red[0] + red[1] + red[2] + red[3];
    float rs = rsqrtf(tot / (float)HD_ + EPS_);

    float xn = val * rs * W[d];
    float xo = oth * rs * W[pd];

    float p  = (float)pos[t];
    int   i  = d & 63;
    float invf = __powf(1000000.0f, -(float)i / 64.0f);
    float ang  = p * invf;
    float c = __cosf(ang), s = __sinf(ang);
    float rot = (d < 64) ? -xo : xo;
    X[base + d] = xn * c + rot * s;
}

// ---------------------------------------------------------------------------
// bf16-WMMA GEMM with TDM-fed, pre-tiled B:  C[M, 128*gridDim.x] = A * B (+Res)
// block = 256 threads (8 waves), tile 64(M) x 128(N), BK = 32, double-buffered.
// B tiles are contiguous 8KB blocks in fragment order; wave 0 drives the TDM.
// ---------------------------------------------------------------------------
__global__ void gemm_bf16_wt_kernel(const float* __restrict__ A, int lda,
                                    const unsigned short* __restrict__ Bt,
                                    const float* __restrict__ Res,
                                    float* __restrict__ C, int ldc,
                                    int M, int K) {
    __shared__ alignas(16) unsigned short As[2][64 * 32];
    __shared__ alignas(16) unsigned short Bs[2][128 * 32];   // [n][k]

    int tid = threadIdx.x;
    int wid = tid >> 5, lane = tid & 31, half = lane >> 4, ln = lane & 15;
    int waveM = (wid >> 2) * 32, waveN = (wid & 3) * 32;
    int bm0 = blockIdx.y * 64, bn0 = blockIdx.x * 128;

    v8f acc[2][2] = {};

    int ar = tid >> 2, ac = (tid & 3) * 8;      // A staging: 8 elems/thread
    int gr = bm0 + ar;
    bool aok = gr < M;
    const float* arow = A + (size_t)gr * lda + ac;

    int nk = K >> 5;
    const unsigned short* btile = Bt + (size_t)blockIdx.x * nk * 4096;
    unsigned bs_lds0 = (unsigned)(size_t)&Bs[0][0];
    unsigned bs_lds1 = (unsigned)(size_t)&Bs[1][0];

    // prologue: stage k-tile 0
    store_pk8(&As[0][ar * 32 + ac], arow, aok);
    if (wid == 0) {
        tdm_load_tile_8kb(bs_lds0, btile);
        __builtin_amdgcn_s_wait_tensorcnt(0);
    }
    __syncthreads();

    for (int it = 0; it < nk; ++it) {
        int cur = it & 1;
        int kk = it << 5;

        FragBF a0, a1, b0, b1;
        LOAD_FRAG(a0, As[cur], (waveM + ln) * 32);
        LOAD_FRAG(a1, As[cur], (waveM + 16 + ln) * 32);
        LOAD_FRAG(b0, Bs[cur], (waveN + ln) * 32);
        LOAD_FRAG(b1, Bs[cur], (waveN + 16 + ln) * 32);

        if (it + 1 < nk) {
            store_pk8(&As[cur ^ 1][ar * 32 + ac], arow + kk + 32, aok);
            if (wid == 0)
                tdm_load_tile_8kb(cur ? bs_lds0 : bs_lds1,
                                  btile + (size_t)(it + 1) * 4096);
            if (it + 2 < nk) {
                __builtin_prefetch((const void*)(arow + kk + 64), 0, 1);
                __builtin_prefetch((const void*)(btile + (size_t)(it + 2) * 4096 + tid * 16), 0, 1);
            }
        }

        acc[0][0] = WMMA_BF16(a0.v, b0.v, acc[0][0]);
        acc[0][1] = WMMA_BF16(a0.v, b1.v, acc[0][1]);
        acc[1][0] = WMMA_BF16(a1.v, b0.v, acc[1][0]);
        acc[1][1] = WMMA_BF16(a1.v, b1.v, acc[1][1]);

        if (wid == 0 && it + 1 < nk)
            __builtin_amdgcn_s_wait_tensorcnt(0);
        __syncthreads();
    }

    #pragma unroll
    for (int ti = 0; ti < 2; ++ti)
        #pragma unroll
        for (int tj = 0; tj < 2; ++tj)
            #pragma unroll
            for (int r = 0; r < 8; ++r) {
                int row = bm0 + waveM + ti * 16 + r + 8 * half;
                int col = bn0 + waveN + tj * 16 + ln;
                if (row < M) {
                    float v = acc[ti][tj][r];
                    if (Res) v += Res[(size_t)row * ldc + col];
                    C[(size_t)row * ldc + col] = v;
                }
            }
}

// ---------------------------------------------------------------------------
// Flash attention (causal, GQA rep=8).  block = 128 threads = 4 waves.
// Each block: one head, 64 query rows (16 per wave). K/V tiles of 32.
// ---------------------------------------------------------------------------
__global__ void attn_kernel(const float* __restrict__ Q,
                            const float* __restrict__ Kb,
                            const float* __restrict__ Vb,
                            float* __restrict__ O) {
    __shared__ alignas(16) unsigned short Qs[64 * 128];
    __shared__ alignas(16) unsigned short Ks[32 * 128];   // [key][d]
    __shared__ alignas(16) unsigned short Vs[128 * 32];   // [d][key] (transposed)
    __shared__ alignas(16) unsigned short Ps[4 * 16 * 32];

    int h  = blockIdx.y;
    int q0 = blockIdx.x * 64;
    int hk = h / (H_ / HK_);
    int tid = threadIdx.x;
    int wid = tid >> 5, lane = tid & 31, half = lane >> 4, ln = lane & 15;

    {   // stage Q tile 64 x 128 -> bf16  (8 x ds_store_b128 per thread)
        int r = tid >> 1, c0 = (tid & 1) * 64;
        const float* qp = Q + ((size_t)(q0 + r) * H_ + h) * HD_ + c0;
        #pragma unroll
        for (int i = 0; i < 8; ++i)
            store_pk8(&Qs[r * 128 + c0 + 8 * i], qp + 8 * i, true);
    }
    __syncthreads();

    v8f oacc[8] = {};
    float mrow[8], lrow[8];
    #pragma unroll
    for (int r = 0; r < 8; ++r) { mrow[r] = -1e30f; lrow[r] = 0.f; }

    const float sc = 0.08838834764831845f;   // 1/sqrt(128)
    const size_t kvstride = (size_t)HK_ * HD_;
    int nkb = (q0 + 64) / 32;                // causal: keys 0 .. q0+63

    for (int kb = 0; kb < nkb; ++kb) {
        int k0 = kb * 32;
        __syncthreads();
        {   // stage K (32x128) row-major
            int r = tid >> 2, c0 = (tid & 3) * 32;
            const float* kp = Kb + ((size_t)(k0 + r) * HK_ + hk) * HD_ + c0;
            #pragma unroll
            for (int i = 0; i < 4; ++i)
                store_pk8(&Ks[r * 128 + c0 + 8 * i], kp + 8 * i, true);
        }
        {   // stage V transposed [d][key], 4 keys packed per ds_store_b64
            int kq = tid >> 4;            // 0..7 -> keys 4kq..4kq+3
            int d0 = (tid & 15) * 8;      // 8 d-values
            const float* vp = Vb + ((size_t)(k0 + 4 * kq) * HK_ + hk) * HD_ + d0;
            float4 a0 = *(const float4*)(vp);
            float4 a1 = *(const float4*)(vp + 4);
            float4 b0 = *(const float4*)(vp + kvstride);
            float4 b1 = *(const float4*)(vp + kvstride + 4);
            float4 c0 = *(const float4*)(vp + 2 * kvstride);
            float4 c1 = *(const float4*)(vp + 2 * kvstride + 4);
            float4 e0 = *(const float4*)(vp + 3 * kvstride);
            float4 e1 = *(const float4*)(vp + 3 * kvstride + 4);
            unsigned short* vb = &Vs[d0 * 32 + 4 * kq];
            *(uint2*)(vb + 0 * 32) = make_uint2(pk2(a0.x, b0.x), pk2(c0.x, e0.x));
            *(uint2*)(vb + 1 * 32) = make_uint2(pk2(a0.y, b0.y), pk2(c0.y, e0.y));
            *(uint2*)(vb + 2 * 32) = make_uint2(pk2(a0.z, b0.z), pk2(c0.z, e0.z));
            *(uint2*)(vb + 3 * 32) = make_uint2(pk2(a0.w, b0.w), pk2(c0.w, e0.w));
            *(uint2*)(vb + 4 * 32) = make_uint2(pk2(a1.x, b1.x), pk2(c1.x, e1.x));
            *(uint2*)(vb + 5 * 32) = make_uint2(pk2(a1.y, b1.y), pk2(c1.y, e1.y));
            *(uint2*)(vb + 6 * 32) = make_uint2(pk2(a1.z, b1.z), pk2(c1.z, e1.z));
            *(uint2*)(vb + 7 * 32) = make_uint2(pk2(a1.w, b1.w), pk2(c1.w, e1.w));
        }
        __syncthreads();

        #pragma unroll
        for (int t = 0; t < 2; ++t) {
            v8f s = {};
            #pragma unroll
            for (int dk = 0; dk < 4; ++dk) {
                FragBF a, b;
                LOAD_FRAG(a, Qs, (wid * 16 + ln) * 128 + dk * 32);
                LOAD_FRAG(b, Ks, (t * 16 + ln) * 128 + dk * 32);
                s = WMMA_BF16(a.v, b.v, s);
            }
            int col = k0 + t * 16 + ln;
            #pragma unroll
            for (int r = 0; r < 8; ++r) {
                int rowg = q0 + wid * 16 + r + 8 * half;
                float sv = s[r] * sc;
                if (col > rowg) sv = -1e30f;
                float mx = sv;
                mx = fmaxf(mx, __shfl_xor(mx, 1, 16));
                mx = fmaxf(mx, __shfl_xor(mx, 2, 16));
                mx = fmaxf(mx, __shfl_xor(mx, 4, 16));
                mx = fmaxf(mx, __shfl_xor(mx, 8, 16));
                float mnew  = fmaxf(mrow[r], mx);
                float alpha = __expf(mrow[r] - mnew);
                mrow[r] = mnew;
                float p  = __expf(sv - mnew);
                float rs = p;
                rs += __shfl_xor(rs, 1, 16);
                rs += __shfl_xor(rs, 2, 16);
                rs += __shfl_xor(rs, 4, 16);
                rs += __shfl_xor(rs, 8, 16);
                lrow[r] = lrow[r] * alpha + rs;
                #pragma unroll
                for (int nd = 0; nd < 8; ++nd) oacc[nd][r] *= alpha;
                Ps[wid * 512 + (r + 8 * half) * 32 + t * 16 + ln] = f2bf(p);
            }
        }

        // P (16x32) @ V (32x128) -> 8 output tiles
        FragBF pa;
        LOAD_FRAG(pa, Ps, wid * 512 + ln * 32);
        #pragma unroll
        for (int nd = 0; nd < 8; ++nd) {
            FragBF vb;
            LOAD_FRAG(vb, Vs, (nd * 16 + ln) * 32);
            oacc[nd] = WMMA_BF16(pa.v, vb.v, oacc[nd]);
        }
    }

    #pragma unroll
    for (int r = 0; r < 8; ++r) {
        float inv = 1.0f / lrow[r];
        int rowg = q0 + wid * 16 + r + 8 * half;
        #pragma unroll
        for (int nd = 0; nd < 8; ++nd) {
            int col = nd * 16 + ln;
            O[((size_t)rowg * H_ + h) * HD_ + col] = oacc[nd][r] * inv;
        }
    }
}

// ---------------------------------------------------------------------------
// Router: logits = x @ router_w, softmax, top-4, renormalize
// ---------------------------------------------------------------------------
__global__ void router_topk_kernel(const float* __restrict__ Xp,
                                   const float* __restrict__ RW,
                                   int* __restrict__ tki,
                                   float* __restrict__ tkw) {
    int t = blockIdx.x, tid = threadIdx.x;
    float acc[E_];
    #pragma unroll
    for (int e = 0; e < E_; ++e) acc[e] = 0.f;
    for (int d = tid; d < D_; d += 256) {
        float xv = Xp[(size_t)t * D_ + d];
        const float* r = RW + (size_t)d * E_;
        #pragma unroll
        for (int e = 0; e < E_; ++e) acc[e] += xv * r[e];
    }
    __shared__ float part[8][E_];
    int wid = tid >> 5, lane = tid & 31;
    #pragma unroll
    for (int e = 0; e < E_; ++e) {
        float v = acc[e];
        #pragma unroll
        for (int o = 16; o > 0; o >>= 1) v += __shfl_xor(v, o, 32);
        if (lane == 0) part[wid][e] = v;
    }
    __syncthreads();
    __shared__ float logits[E_];
    if (tid < E_) {
        float s = 0.f;
        #pragma unroll
        for (int w = 0; w < 8; ++w) s += part[w][tid];
        logits[tid] = s;
    }
    __syncthreads();
    if (tid == 0) {
        float m = -1e30f;
        for (int e = 0; e < E_; ++e) m = fmaxf(m, logits[e]);
        float p[E_], se = 0.f;
        for (int e = 0; e < E_; ++e) { p[e] = __expf(logits[e] - m); se += p[e]; }
        float inv = 1.f / se;
        for (int e = 0; e < E_; ++e) p[e] *= inv;
        float tw[TOPK_]; int ti[TOPK_]; float tot = 0.f;
        for (int j = 0; j < TOPK_; ++j) {
            float best = -1.f; int bi = 0;
            for (int e = 0; e < E_; ++e)
                if (p[e] > best) { best = p[e]; bi = e; }
            p[bi] = -2.f;
            tw[j] = best; ti[j] = bi; tot += best;
        }
        float itot = 1.f / tot;
        for (int j = 0; j < TOPK_; ++j) {
            tki[t * TOPK_ + j] = ti[j];
            tkw[t * TOPK_ + j] = tw[j] * itot;
        }
    }
}

__global__ void count_kernel(const int* __restrict__ tki, int* __restrict__ cnt) {
    int i = blockIdx.x * blockDim.x + threadIdx.x;
    if (i < S_ * TOPK_) atomicAdd(&cnt[tki[i]], 1);
}

__global__ void offsets_kernel(const int* __restrict__ cnt, int* __restrict__ ofs) {
    if (threadIdx.x == 0) {
        int run = 0;
        for (int e = 0; e < E_; ++e) { ofs[e] = run; run += cnt[e]; }
    }
}

__global__ void fill_kernel(const int* __restrict__ tki, const float* __restrict__ tkw,
                            const int* __restrict__ ofs, int* __restrict__ cur,
                            int* __restrict__ rowIdx, float* __restrict__ rowW) {
    int i = blockIdx.x * blockDim.x + threadIdx.x;
    if (i < S_ * TOPK_) {
        int e = tki[i];
        int p = atomicAdd(&cur[e], 1);
        int s = ofs[e] + p;
        rowIdx[s] = i >> 2;       // token
        rowW[s]   = tkw[i];
    }
}

// ---------------------------------------------------------------------------
// MoE GEMM 1: act = silu(Xe @ gate_w[e]) * (Xe @ up_w[e]), token-gathered
// grid: (tokenTile<=32, F/128, E), double-buffered LDS
// ---------------------------------------------------------------------------
__global__ void moe_gemm1_kernel(const float* __restrict__ Xp,
                                 const float* __restrict__ GW,
                                 const float* __restrict__ UW,
                                 const int* __restrict__ cnt,
                                 const int* __restrict__ ofs,
                                 const int* __restrict__ rowIdx,
                                 float* __restrict__ Act) {
    int e  = blockIdx.z;
    int nT = cnt[e];
    int m0 = blockIdx.x * 64;
    if (m0 >= nT) return;
    int base = ofs[e];

    __shared__ int rows[64];
    __shared__ alignas(16) unsigned short As[2][64 * 32];
    __shared__ alignas(16) unsigned short BsG[2][128 * 32];
    __shared__ alignas(16) unsigned short BsU[2][128 * 32];

    int tid = threadIdx.x;
    if (tid < 64) rows[tid] = (m0 + tid < nT) ? rowIdx[base + m0 + tid] : -1;
    __syncthreads();

    int wid = tid >> 5, lane = tid & 31, half = lane >> 4, ln = lane & 15;
    int waveM = (wid >> 2) * 32, waveN = (wid & 3) * 32;
    int bn0 = blockIdx.y * 128;

    const float* G = GW + (size_t)e * D_ * F_ + bn0;
    const float* U = UW + (size_t)e * D_ * F_ + bn0;

    v8f accG[2][2] = {}, accU[2][2] = {};

    int ar = tid >> 2, ac = (tid & 3) * 8;
    int rrow = rows[ar];
    bool aok = rrow >= 0;
    const float* arow = Xp + (size_t)(aok ? rrow : 0) * D_ + ac;

    store_pk8(&As[0][ar * 32 + ac], arow, aok);
    stage_bT(BsG[0], G, F_, tid);
    stage_bT(BsU[0], U, F_, tid);
    __syncthreads();

    int nk = D_ >> 5;
    for (int it = 0; it < nk; ++it) {
        int cur = it & 1;
        int kk = it << 5;

        FragBF a0, a1, g0, g1, u0, u1;
        LOAD_FRAG(a0, As[cur], (waveM + ln) * 32);
        LOAD_FRAG(a1, As[cur], (waveM + 16 + ln) * 32);
        LOAD_FRAG(g0, BsG[cur], (waveN + ln) * 32);
        LOAD_FRAG(g1, BsG[cur], (waveN + 16 + ln) * 32);
        LOAD_FRAG(u0, BsU[cur], (waveN + ln) * 32);
        LOAD_FRAG(u1, BsU[cur], (waveN + 16 + ln) * 32);

        if (it + 1 < nk) {
            store_pk8(&As[cur ^ 1][ar * 32 + ac], arow + kk + 32, aok);
            stage_bT(BsG[cur ^ 1], G + (size_t)(kk + 32) * F_, F_, tid);
            stage_bT(BsU[cur ^ 1], U + (size_t)(kk + 32) * F_, F_, tid);
            if (it + 2 < nk) {
                __builtin_prefetch((const void*)(arow + kk + 64), 0, 1);
                __builtin_prefetch((const void*)(G + (size_t)(kk + 64) * F_ + tid), 0, 1);
                __builtin_prefetch((const void*)(U + (size_t)(kk + 64) * F_ + tid), 0, 1);
            }
        }

        accG[0][0] = WMMA_BF16(a0.v, g0.v, accG[0][0]);
        accG[0][1] = WMMA_BF16(a0.v, g1.v, accG[0][1]);
        accG[1][0] = WMMA_BF16(a1.v, g0.v, accG[1][0]);
        accG[1][1] = WMMA_BF16(a1.v, g1.v, accG[1][1]);
        accU[0][0] = WMMA_BF16(a0.v, u0.v, accU[0][0]);
        accU[0][1] = WMMA_BF16(a0.v, u1.v, accU[0][1]);
        accU[1][0] = WMMA_BF16(a1.v, u0.v, accU[1][0]);
        accU[1][1] = WMMA_BF16(a1.v, u1.v, accU[1][1]);
        __syncthreads();
    }

    #pragma unroll
    for (int ti = 0; ti < 2; ++ti)
        #pragma unroll
        for (int tj = 0; tj < 2; ++tj)
            #pragma unroll
            for (int r = 0; r < 8; ++r) {
                int m = waveM + ti * 16 + r + 8 * half;
                if (m0 + m < nT) {
                    int col = bn0 + waveN + tj * 16 + ln;
                    float g = accG[ti][tj][r];
                    float u = accU[ti][tj][r];
                    float a = g * (1.0f / (1.0f + __expf(-g))) * u;
                    Act[(size_t)(base + m0 + m) * F_ + col] = a;
                }
            }
}

// ---------------------------------------------------------------------------
// MoE GEMM 2: out[token] += (act @ down_w[e]) * weight     (atomic scatter)
// grid: (tokenTile<=32, D/128, E), double-buffered LDS
// ---------------------------------------------------------------------------
__global__ void moe_gemm2_kernel(const float* __restrict__ Act,
                                 const float* __restrict__ DW,
                                 const int* __restrict__ cnt,
                                 const int* __restrict__ ofs,
                                 const int* __restrict__ rowIdx,
                                 const float* __restrict__ rowW,
                                 float* __restrict__ Out) {
    int e  = blockIdx.z;
    int nT = cnt[e];
    int m0 = blockIdx.x * 64;
    if (m0 >= nT) return;
    int base = ofs[e];

    __shared__ alignas(16) unsigned short As[2][64 * 32];
    __shared__ alignas(16) unsigned short Bs[2][128 * 32];

    int tid = threadIdx.x;
    int wid = tid >> 5, lane = tid & 31, half = lane >> 4, ln = lane & 15;
    int waveM = (wid >> 2) * 32, waveN = (wid & 3) * 32;
    int bn0 = blockIdx.y * 128;

    const float* B = DW + (size_t)e * F_ * D_ + bn0;

    v8f acc[2][2] = {};

    int ar = tid >> 2, ac = (tid & 3) * 8;
    bool aok = (m0 + ar < nT);
    const float* arow = Act + (size_t)(base + m0 + ar) * F_ + ac;

    store_pk8(&As[0][ar * 32 + ac], arow, aok);
    stage_bT(Bs[0], B, D_, tid);
    __syncthreads();

    int nk = F_ >> 5;
    for (int it = 0; it < nk; ++it) {
        int cur = it & 1;
        int kk = it << 5;

        FragBF a0, a1, b0, b1;
        LOAD_FRAG(a0, As[cur], (waveM + ln) * 32);
        LOAD_FRAG(a1, As[cur], (waveM + 16 + ln) * 32);
        LOAD_FRAG(b0, Bs[cur], (waveN + ln) * 32);
        LOAD_FRAG(b1, Bs[cur], (waveN + 16 + ln) * 32);

        if (it + 1 < nk) {
            store_pk8(&As[cur ^ 1][ar * 32 + ac], arow + kk + 32, aok);
            stage_bT(Bs[cur ^ 1], B + (size_t)(kk + 32) * D_, D_, tid);
            if (it + 2 < nk) {
                __builtin_prefetch((const void*)(arow + kk + 64), 0, 1);
                __builtin_prefetch((const void*)(B + (size_t)(kk + 64) * D_ + tid), 0, 1);
            }
        }

        acc[0][0] = WMMA_BF16(a0.v, b0.v, acc[0][0]);
        acc[0][1] = WMMA_BF16(a0.v, b1.v, acc[0][1]);
        acc[1][0] = WMMA_BF16(a1.v, b0.v, acc[1][0]);
        acc[1][1] = WMMA_BF16(a1.v, b1.v, acc[1][1]);
        __syncthreads();
    }

    #pragma unroll
    for (int ti = 0; ti < 2; ++ti)
        #pragma unroll
        for (int tj = 0; tj < 2; ++tj)
            #pragma unroll
            for (int r = 0; r < 8; ++r) {
                int m = waveM + ti * 16 + r + 8 * half;
                if (m0 + m < nT) {
                    int   token = rowIdx[base + m0 + m];
                    float w     = rowW[base + m0 + m];
                    int   col   = bn0 + waveN + tj * 16 + ln;
                    atomicAdd(&Out[(size_t)token * D_ + col], acc[ti][tj][r] * w);
                }
            }
}

// ---------------------------------------------------------------------------
// Host driver
// ---------------------------------------------------------------------------
extern "C" void kernel_launch(void* const* d_in, const int* in_sizes, int n_in,
                              void* d_out, int out_size, void* d_ws, size_t ws_size,
                              hipStream_t stream) {
    (void)in_sizes; (void)n_in; (void)out_size; (void)ws_size;

    const float* hs        = (const float*)d_in[0];
    const int*   pos       = (const int*)d_in[1];
    const float* ln_in_w   = (const float*)d_in[2];
    const float* q_w       = (const float*)d_in[3];
    const float* k_w       = (const float*)d_in[4];
    const float* v_w       = (const float*)d_in[5];
    const float* o_w       = (const float*)d_in[6];
    const float* q_ln_w    = (const float*)d_in[7];
    const float* k_ln_w    = (const float*)d_in[8];
    const float* ln_post_w = (const float*)d_in[9];
    const float* router_w  = (const float*)d_in[10];
    const float* gate_w    = (const float*)d_in[11];
    const float* up_w      = (const float*)d_in[12];
    const float* down_w    = (const float*)d_in[13];
    float* out = (float*)d_out;

    char* ws = (char*)d_ws;
    size_t off = 0;
    auto alloc = [&](size_t bytes) -> void* {
        void* p = ws + off;
        off += (bytes + 255) & ~(size_t)255;
        return p;
    };
    float* xnorm  = (float*)alloc((size_t)S_ * D_ * 4);
    float* qb     = (float*)alloc((size_t)S_ * H_ * HD_ * 4);
    float* kb     = (float*)alloc((size_t)S_ * HK_ * HD_ * 4);
    float* vb     = (float*)alloc((size_t)S_ * HK_ * HD_ * 4);
    float* ctx    = (float*)alloc((size_t)S_ * H_ * HD_ * 4);
    float* xpost  = (float*)alloc((size_t)S_ * D_ * 4);
    int*   tki    = (int*)alloc((size_t)S_ * TOPK_ * 4);
    float* tkw    = (float*)alloc((size_t)S_ * TOPK_ * 4);
    int*   cnt    = (int*)alloc(E_ * 4);
    int*   ofs    = (int*)alloc(E_ * 4);
    int*   cur    = (int*)alloc(E_ * 4);
    int*   rowIdx = (int*)alloc((size_t)S_ * TOPK_ * 4);
    float* rowW   = (float*)alloc((size_t)S_ * TOPK_ * 4);
    float* act    = (float*)alloc((size_t)S_ * TOPK_ * F_ * 4);
    unsigned short* qwT = (unsigned short*)alloc((size_t)D_ * H_ * HD_ * 2);
    unsigned short* kwT = (unsigned short*)alloc((size_t)D_ * HK_ * HD_ * 2);
    unsigned short* vwT = (unsigned short*)alloc((size_t)D_ * HK_ * HD_ * 2);
    unsigned short* owT = (unsigned short*)alloc((size_t)H_ * HD_ * D_ * 2);

    // 0. pre-tile projection weights to bf16 fragment layout (TDM source)
    tile_weights_kernel<<<dim3(D_ / 32, (H_ * HD_) / 128), 256, 0, stream>>>(
        q_w, qwT, D_, H_ * HD_);
    tile_weights_kernel<<<dim3(D_ / 32, (HK_ * HD_) / 128), 256, 0, stream>>>(
        k_w, kwT, D_, HK_ * HD_);
    tile_weights_kernel<<<dim3(D_ / 32, (HK_ * HD_) / 128), 256, 0, stream>>>(
        v_w, vwT, D_, HK_ * HD_);
    tile_weights_kernel<<<dim3((H_ * HD_) / 32, D_ / 128), 256, 0, stream>>>(
        o_w, owT, H_ * HD_, D_);

    // 1. input RMSNorm
    rmsnorm_kernel<<<S_, 256, 0, stream>>>(hs, ln_in_w, xnorm, D_);

    // 2. Q/K/V projections (bf16 WMMA, TDM-fed B tiles)
    gemm_bf16_wt_kernel<<<dim3((H_ * HD_) / 128, S_ / 64), 256, 0, stream>>>(
        xnorm, D_, qwT, nullptr, qb, H_ * HD_, S_, D_);
    gemm_bf16_wt_kernel<<<dim3((HK_ * HD_) / 128, S_ / 64), 256, 0, stream>>>(
        xnorm, D_, kwT, nullptr, kb, HK_ * HD_, S_, D_);
    gemm_bf16_wt_kernel<<<dim3((HK_ * HD_) / 128, S_ / 64), 256, 0, stream>>>(
        xnorm, D_, vwT, nullptr, vb, HK_ * HD_, S_, D_);

    // 3. QK-norm + RoPE
    qknorm_rope_kernel<<<dim3(S_, H_), 128, 0, stream>>>(qb, q_ln_w, pos, H_);
    qknorm_rope_kernel<<<dim3(S_, HK_), 128, 0, stream>>>(kb, k_ln_w, pos, HK_);

    // 4. causal GQA flash attention
    attn_kernel<<<dim3(S_ / 64, H_), 128, 0, stream>>>(qb, kb, vb, ctx);

    // 5. output projection + residual -> d_out
    gemm_bf16_wt_kernel<<<dim3(D_ / 128, S_ / 64), 256, 0, stream>>>(
        ctx, H_ * HD_, owT, hs, out, D_, S_, H_ * HD_);

    // 6. post RMSNorm
    rmsnorm_kernel<<<S_, 256, 0, stream>>>(out, ln_post_w, xpost, D_);

    // 7. router + top-k
    router_topk_kernel<<<S_, 256, 0, stream>>>(xpost, router_w, tki, tkw);

    // 8. token -> expert compaction
    (void)hipMemsetAsync(cnt, 0, E_ * 4, stream);
    (void)hipMemsetAsync(cur, 0, E_ * 4, stream);
    count_kernel<<<(S_ * TOPK_) / 256, 256, 0, stream>>>(tki, cnt);
    offsets_kernel<<<1, 32, 0, stream>>>(cnt, ofs);
    fill_kernel<<<(S_ * TOPK_) / 256, 256, 0, stream>>>(tki, tkw, ofs, cur, rowIdx, rowW);

    // 9. expert GEMMs (gathered), scatter-add into d_out
    moe_gemm1_kernel<<<dim3(S_ / 64, F_ / 128, E_), 256, 0, stream>>>(
        xpost, gate_w, up_w, cnt, ofs, rowIdx, act);
    moe_gemm2_kernel<<<dim3(S_ / 64, D_ / 128, E_), 256, 0, stream>>>(
        act, down_w, cnt, ofs, rowIdx, rowW, out);
}
